// SelfAttention_51075751084749
// MI455X (gfx1250) — matrix operand order repaired and checked by reading, not compiled, not run on hardware
//
#include <hip/hip_runtime.h>
#include <hip/hip_bf16.h>

// ---------------------------------------------------------------------------
// Self-attention (QKV proj + softmax(QK^T)V), S=8192, D=2048, gfx1250.
// All GEMMs via v_wmma_f32_16x16x32_bf16 (f32 accumulate); softmax in f32.
// B operands stored transposed [N][K] -> every fragment load is a contiguous
// b128 pair. global_prefetch_b8 used to stage the next K tile.
// ---------------------------------------------------------------------------

typedef __attribute__((ext_vector_type(16))) __bf16 v16bf;
typedef __attribute__((ext_vector_type(8)))  __bf16 v8bf;
typedef __attribute__((ext_vector_type(8)))  float  v8f;

#define SEQ    8192
#define DMODEL 2048
#define N3     6144                       // 3*DMODEL, qkv row stride
#define SCALE  0.022097086912079608f      // 1/sqrt(2048)

__device__ __forceinline__ v8f wmma_bf16(v16bf a, v16bf b, v8f c) {
  return __builtin_amdgcn_wmma_f32_16x16x32_bf16(
      false, a, false, b, (short)0, c, false, false);
}

// A-fragment: 16x32 bf16 MxK from row-major [M][K] (ld).
// Lane m (0..15): K={0..7,16..23}; lane 16+m: K={8..15,24..31}.
__device__ __forceinline__ v16bf load_frag_a(const __bf16* base, int ld) {
  int lane = threadIdx.x & 31;
  int m  = lane & 15;
  int kh = (lane >> 4) << 3;              // 0 or 8
  const __bf16* p = base + (size_t)m * ld + kh;
  v8bf lo = *(const v8bf*)(p);
  v8bf hi = *(const v8bf*)(p + 16);
  return __builtin_shufflevector(lo, hi, 0,1,2,3,4,5,6,7,8,9,10,11,12,13,14,15);
}

// B-fragment: 32x16 bf16 KxN, sourced from TRANSPOSED storage [N][K] (ld).
// Lane n (0..15): K={0..7,16..23}; lane 16+n: K={8..15,24..31}.
__device__ __forceinline__ v16bf load_frag_b(const __bf16* base, int ld) {
  int lane = threadIdx.x & 31;
  int n  = lane & 15;
  int k0 = (lane >> 4) << 3;              // 0 or 8
  const __bf16* p = base + (size_t)n * ld + k0;
  v8bf lo = *(const v8bf*)(p);
  v8bf hi = *(const v8bf*)(p + 16);
  return __builtin_shufflevector(lo, hi, 0,1,2,3,4,5,6,7,8,9,10,11,12,13,14,15);
}

// ---------------------------------------------------------------------------
// Kernel 1: f32 -> bf16 (x)
// ---------------------------------------------------------------------------
__global__ void cvt_f32_bf16(const float* __restrict__ in,
                             __bf16* __restrict__ out, long n) {
  long i = (long)blockIdx.x * blockDim.x + threadIdx.x;
  long stride = (long)gridDim.x * blockDim.x;
  for (; i < n; i += stride) out[i] = (__bf16)in[i];
}

// ---------------------------------------------------------------------------
// Kernel 2: W [D][N3] f32 -> Wt [N3][D] bf16 (convert + transpose, LDS tile)
// ---------------------------------------------------------------------------
__global__ __launch_bounds__(256)
void cvt_transpose_W(const float* __restrict__ W, __bf16* __restrict__ Wt) {
  __shared__ float tile[32][33];
  int bx = blockIdx.x * 32;               // N3 dim
  int by = blockIdx.y * 32;               // D dim
  int tx = threadIdx.x & 31, ty = threadIdx.x >> 5;
#pragma unroll
  for (int i = ty; i < 32; i += 8)
    tile[i][tx] = W[(size_t)(by + i) * N3 + bx + tx];
  __syncthreads();
#pragma unroll
  for (int i = ty; i < 32; i += 8)
    Wt[(size_t)(bx + i) * DMODEL + by + tx] = (__bf16)tile[tx][i];
}

// ---------------------------------------------------------------------------
// Kernel 4: V slice of qkv [SEQ][D] (stride N3) -> Vt [D][SEQ] bf16
// ---------------------------------------------------------------------------
__global__ __launch_bounds__(256)
void transpose_v(const __bf16* __restrict__ qkv, __bf16* __restrict__ Vt) {
  __shared__ __bf16 tile[32][33];
  int bx = blockIdx.x * 32;               // SEQ dim
  int by = blockIdx.y * 32;               // D dim
  int tx = threadIdx.x & 31, ty = threadIdx.x >> 5;
#pragma unroll
  for (int i = ty; i < 32; i += 8)
    tile[i][tx] = qkv[(size_t)(bx + i) * N3 + 2 * DMODEL + by + tx];
  __syncthreads();
#pragma unroll
  for (int i = ty; i < 32; i += 8)
    Vt[(size_t)(by + i) * SEQ + bx + tx] = tile[tx][i];
}

// ---------------------------------------------------------------------------
// Kernel 3/7: tiled bf16 GEMM, f32 accumulate, B pre-transposed.
//   C[M,N] = A[M,K](lda) @ Bt[N,K](ldbt)^T (+bias)
//   256 threads = 8 waves (2x4); WG tile 64x256; wave tile 32x64; K tile 64.
//   LDS: As 64x64 (8KB) + Bs 256x64 (32KB) = 40KB, all b128 traffic.
//   Next-tile global_prefetch_b8 issued before the compute phase.
// ---------------------------------------------------------------------------
template <bool HAS_BIAS, bool OUT_BF16>
__global__ __launch_bounds__(256)
void gemm_bf16(const __bf16* __restrict__ A, int lda,
               const __bf16* __restrict__ Bt, int ldbt,
               const float* __restrict__ bias,
               float* __restrict__ Cf, __bf16* __restrict__ Cbf, int ldc,
               int K) {
  __shared__ __bf16 As[64 * 64];
  __shared__ __bf16 Bs[256 * 64];

  const int t    = threadIdx.x;
  const int lane = t & 31;
  const int wave = t >> 5;
  const int wr   = wave >> 2;             // 0..1
  const int wc   = wave & 3;              // 0..3
  const int M0   = blockIdx.y * 64;
  const int N0   = blockIdx.x * 256;

  v8f acc[2][4];
#pragma unroll
  for (int i = 0; i < 2; ++i)
#pragma unroll
    for (int j = 0; j < 4; ++j) acc[i][j] = (v8f){};

  // per-thread fill coordinates
  const int arow = t >> 3, ach = t & 7;   // A: 1 chunk at (t>>3, t&7), 2nd below

  for (int kt = 0; kt < K; kt += 64) {
    // ---- A tile 64x64: 2 b128 chunks per thread
#pragma unroll
    for (int i = 0; i < 2; ++i) {
      int cid = t + i * 256;
      int row = cid >> 3, ch = cid & 7;
      *(v8bf*)&As[row * 64 + ch * 8] =
          *(const v8bf*)&A[(size_t)(M0 + row) * lda + kt + ch * 8];
    }
    // ---- B tile 256x64 (already [N][K]): 8 b128 chunks per thread
#pragma unroll
    for (int i = 0; i < 8; ++i) {
      int cid = t + i * 256;
      int n = cid >> 3, ch = cid & 7;
      *(v8bf*)&Bs[n * 64 + ch * 8] =
          *(const v8bf*)&Bt[(size_t)(N0 + n) * ldbt + kt + ch * 8];
    }
    // ---- prefetch next K tile (speculative, no counters)
    if (kt + 64 < K) {
      __builtin_prefetch(&A[(size_t)(M0 + arow) * lda + kt + 64 + ach * 8], 0, 1);
#pragma unroll
      for (int i = 0; i < 8; i += 2) {
        int cid = t + i * 256;
        int n = cid >> 3, ch = cid & 7;
        __builtin_prefetch(&Bt[(size_t)(N0 + n) * ldbt + kt + 64 + ch * 8], 0, 1);
      }
    }
    __syncthreads();

#pragma unroll
    for (int ks = 0; ks < 64; ks += 32) {
      v16bf af[2], bfg[4];
#pragma unroll
      for (int i = 0; i < 2; ++i)
        af[i] = load_frag_a(&As[(wr * 32 + i * 16) * 64 + ks], 64);
#pragma unroll
      for (int j = 0; j < 4; ++j)
        bfg[j] = load_frag_b(&Bs[(wc * 64 + j * 16) * 64 + ks], 64);
#pragma unroll
      for (int i = 0; i < 2; ++i)
#pragma unroll
        for (int j = 0; j < 4; ++j)
          acc[i][j] = wmma_bf16(af[i], bfg[j], acc[i][j]);
    }
    __syncthreads();
  }

  // ---- branch-free epilogue; bias loaded once per output column
  const int hi = lane >> 4;
  const int cn = lane & 15;
#pragma unroll
  for (int j = 0; j < 4; ++j) {
    const int col = N0 + wc * 64 + j * 16 + cn;
    const float bv = HAS_BIAS ? bias[col] : 0.0f;
#pragma unroll
    for (int i = 0; i < 2; ++i)
#pragma unroll
      for (int r = 0; r < 8; ++r) {
        int row = M0 + wr * 32 + i * 16 + r + 8 * hi;
        float v = acc[i][j][r] + bv;
        if (OUT_BF16) Cbf[(size_t)row * ldc + col] = (__bf16)v;
        else          Cf [(size_t)row * ldc + col] = v;
      }
  }
}

// ---------------------------------------------------------------------------
// S-tile block: 16 query rows x 64 keys, 4 accumulators sharing one Q frag.
// K matrix [keys][d] row-major is exactly the [N][K] layout for the B loader.
// ---------------------------------------------------------------------------
__device__ __forceinline__ void score_block(const __bf16* Q, const __bf16* Kj,
                                            v8f acc[4]) {
#pragma unroll
  for (int c = 0; c < 4; ++c) acc[c] = (v8f){};
#pragma unroll 2
  for (int kt = 0; kt < DMODEL; kt += 32) {
    v16bf a = load_frag_a(Q + kt, N3);
#pragma unroll
    for (int c = 0; c < 4; ++c)
      acc[c] = wmma_bf16(a, load_frag_b(Kj + (size_t)(c * 16) * N3 + kt, N3),
                         acc[c]);
  }
}

__device__ __forceinline__ void prefetch_keys(const __bf16* Kj) {
  // stage next 64-key block's rows (one line per fragment row this lane reads)
  int lane = threadIdx.x & 31;
  int n  = lane & 15;
  int k0 = (lane >> 4) << 4;
#pragma unroll
  for (int c = 0; c < 4; ++c)
    __builtin_prefetch(Kj + (size_t)(c * 16 + n) * N3 + k0, 0, 1);
}

// ---------------------------------------------------------------------------
// Kernel 5: per-row softmax statistics (online max / sum-exp).
// One wave = 16 query rows; 64 keys per iteration.
// ---------------------------------------------------------------------------
__global__ __launch_bounds__(256)
void attn_stats(const __bf16* __restrict__ qkv, float* __restrict__ stats) {
  const int wave = threadIdx.x >> 5;
  const int lane = threadIdx.x & 31;
  const int row0 = (blockIdx.x * 8 + wave) * 16;
  const __bf16* Q  = qkv + (size_t)row0 * N3;
  const __bf16* Kb = qkv + DMODEL;

  float mrun[8], lrun[8];
#pragma unroll
  for (int r = 0; r < 8; ++r) { mrun[r] = -3.0e30f; lrun[r] = 0.0f; }

#pragma unroll 1
  for (int j = 0; j < SEQ; j += 64) {
    if (j + 64 < SEQ) prefetch_keys(Kb + (size_t)(j + 64) * N3);
    v8f acc[4];
    score_block(Q, Kb + (size_t)j * N3, acc);
#pragma unroll
    for (int r = 0; r < 8; ++r) {
      float v0 = acc[0][r] * SCALE, v1 = acc[1][r] * SCALE;
      float v2 = acc[2][r] * SCALE, v3 = acc[3][r] * SCALE;
      float mx = fmaxf(fmaxf(v0, v1), fmaxf(v2, v3));
#pragma unroll
      for (int off = 1; off < 16; off <<= 1)
        mx = fmaxf(mx, __shfl_xor(mx, off));
      float mnew = fmaxf(mrun[r], mx);
      float p = __expf(v0 - mnew) + __expf(v1 - mnew) +
                __expf(v2 - mnew) + __expf(v3 - mnew);
#pragma unroll
      for (int off = 1; off < 16; off <<= 1)
        p += __shfl_xor(p, off);
      lrun[r] = lrun[r] * __expf(mrun[r] - mnew) + p;
      mrun[r] = mnew;
    }
  }
  if ((lane & 15) == 0) {
    int hi = lane >> 4;
#pragma unroll
    for (int r = 0; r < 8; ++r) {
      int row = row0 + r + 8 * hi;
      stats[2 * row]     = mrun[r];
      stats[2 * row + 1] = lrun[r];
    }
  }
}

// ---------------------------------------------------------------------------
// Kernel 6: P = softmax(scores) bf16 [SEQ][SEQ] (recompute S-tiles).
// ---------------------------------------------------------------------------
__global__ __launch_bounds__(256)
void attn_probs(const __bf16* __restrict__ qkv,
                const float* __restrict__ stats, __bf16* __restrict__ P) {
  const int wave = threadIdx.x >> 5;
  const int lane = threadIdx.x & 31;
  const int row0 = (blockIdx.x * 8 + wave) * 16;
  const int hi = lane >> 4, cn = lane & 15;
  const __bf16* Q  = qkv + (size_t)row0 * N3;
  const __bf16* Kb = qkv + DMODEL;

  float mrow[8], linv[8];
#pragma unroll
  for (int r = 0; r < 8; ++r) {
    int row = row0 + r + 8 * hi;
    mrow[r] = stats[2 * row];
    linv[r] = 1.0f / stats[2 * row + 1];
  }

#pragma unroll 1
  for (int j = 0; j < SEQ; j += 64) {
    if (j + 64 < SEQ) prefetch_keys(Kb + (size_t)(j + 64) * N3);
    v8f acc[4];
    score_block(Q, Kb + (size_t)j * N3, acc);
#pragma unroll
    for (int c = 0; c < 4; ++c)
#pragma unroll
      for (int r = 0; r < 8; ++r) {
        float p = __expf(acc[c][r] * SCALE - mrow[r]) * linv[r];
        P[(size_t)(row0 + r + 8 * hi) * SEQ + j + c * 16 + cn] = (__bf16)p;
      }
  }
}

// ---------------------------------------------------------------------------
// Host launcher. Workspace (256B aligned):
//   xbf [SEQ*D] 32MB | Wt [N3*D] 24MB | qkv [SEQ*N3] 96MB |
//   stats [SEQ*2] 64KB | Vt [D*SEQ] 32MB | P [SEQ*SEQ] 128MB  (~312MB)
// ---------------------------------------------------------------------------
extern "C" void kernel_launch(void* const* d_in, const int* in_sizes, int n_in,
                              void* d_out, int out_size, void* d_ws, size_t ws_size,
                              hipStream_t stream) {
  const float* x = (const float*)d_in[0];
  const float* W = (const float*)d_in[1];
  const float* b = (const float*)d_in[2];
  float* out = (float*)d_out;

  char* ws = (char*)d_ws;
  auto al = [](size_t v) { return (v + 255) & ~(size_t)255; };
  size_t off = 0;
  __bf16* xbf   = (__bf16*)(ws + off); off += al((size_t)SEQ * DMODEL * 2);
  __bf16* Wt    = (__bf16*)(ws + off); off += al((size_t)N3 * DMODEL * 2);
  __bf16* qkvbf = (__bf16*)(ws + off); off += al((size_t)SEQ * N3 * 2);
  float*  stats = (float*) (ws + off); off += al((size_t)SEQ * 2 * 4);
  __bf16* Vt    = (__bf16*)(ws + off); off += al((size_t)DMODEL * SEQ * 2);
  __bf16* Pbf   = (__bf16*)(ws + off); off += al((size_t)SEQ * SEQ * 2);

  // 1) x -> bf16
  cvt_f32_bf16<<<2048, 256, 0, stream>>>(x, xbf, (long)SEQ * DMODEL);
  // 2) W -> Wt (bf16, transposed)
  cvt_transpose_W<<<dim3(N3 / 32, DMODEL / 32), 256, 0, stream>>>(W, Wt);
  // 3) qkv = x @ W + b  (bf16, [SEQ][N3])
  gemm_bf16<true, true><<<dim3(N3 / 256, SEQ / 64), 256, 0, stream>>>(
      xbf, DMODEL, Wt, DMODEL, b, nullptr, qkvbf, N3, DMODEL);
  // 4) Vt = V^T (bf16, [D][SEQ])
  transpose_v<<<dim3(SEQ / 32, DMODEL / 32), 256, 0, stream>>>(qkvbf, Vt);
  // 5) softmax row stats
  attn_stats<<<SEQ / (16 * 8), 256, 0, stream>>>(qkvbf, stats);
  // 6) P = softmax(q k^T * scale)
  attn_probs<<<SEQ / (16 * 8), 256, 0, stream>>>(qkvbf, stats, Pbf);
  // 7) out = P @ V  (f32, [SEQ][D])
  gemm_bf16<false, false><<<dim3(DMODEL / 256, SEQ / 64), 256, 0, stream>>>(
      Pbf, SEQ, Vt, SEQ, nullptr, out, nullptr, DMODEL, SEQ);
}
